// MultiHeadAttentionSelf_2276332667455
// MI455X (gfx1250) — matrix operand order repaired and checked by reading, not compiled, not run on hardware
//
#include <hip/hip_runtime.h>
#include <hip/hip_bf16.h>

using bf16 = __bf16;
typedef __attribute__((ext_vector_type(16))) __bf16 v16bf;
typedef __attribute__((ext_vector_type(8)))  __bf16 v8bf;
typedef __attribute__((ext_vector_type(8)))  float  v8f;
typedef __attribute__((ext_vector_type(4))) unsigned int v4u;
typedef __attribute__((ext_vector_type(8))) int v8i;
typedef __attribute__((ext_vector_type(4))) int v4i;

#define WMMA_BF16(a, b, c) \
  __builtin_amdgcn_wmma_f32_16x16x32_bf16(false, (a), false, (b), (short)0, (c), false, false)

static constexpr int Bb  = 4;
static constexpr int TQ  = 1024;
static constexpr int TKV = 2048;
static constexpr int Dd  = 1024;
static constexpr int DH  = 64;
static constexpr float SCALE = 0.35355339059327373f; // 64^-0.25

#if defined(__has_builtin)
#  if __has_builtin(__builtin_amdgcn_tensor_load_to_lds)
#    define HAVE_TDM 1
#  endif
#  if __has_builtin(__builtin_amdgcn_s_wait_tensorcnt)
#    define HAVE_TENSORCNT 1
#  endif
#endif

// ---------- fragment load helpers ----------
__device__ inline v8bf ld8(const bf16* p) {
  union { uint4 u; v8bf v; } t;
  t.u = *reinterpret_cast<const uint4*>(p);
  return t.v;
}
__device__ inline v16bf combine16(v8bf lo, v8bf hi) {
  v16bf r;
#pragma unroll
  for (int i = 0; i < 8; ++i) { r[i] = lo[i]; r[8 + i] = hi[i]; }
  return r;
}
// A fragment, row-major: elems 0..15 = K kb..kb+15
__device__ inline v16bf load_a16(const bf16* rowp, int akb) {
  return combine16(ld8(rowp + akb), ld8(rowp + akb + 8));
}
// B fragment, 16 contiguous bf16 along K
__device__ inline v16bf load_b16(const bf16* p) {
  return combine16(ld8(p), ld8(p + 8));
}
// B fragment from TDM-written LDS: volatile so reads of compiler-invisible
// (DMA-written) shared memory are never elided/reordered
__device__ inline v16bf load_b16_lds(const bf16* p) {
  const volatile unsigned* q = reinterpret_cast<const volatile unsigned*>(p);
  union { unsigned u[8]; v16bf v; } t;
#pragma unroll
  for (int i = 0; i < 8; ++i) t.u[i] = q[i];
  return t.v;
}
__device__ inline v8f vzero8() {
  v8f z = {0.f, 0.f, 0.f, 0.f, 0.f, 0.f, 0.f, 0.f};
  return z;
}
__device__ inline unsigned lds_off(const void* p) {
  return (unsigned)(uintptr_t)p; // low 32 bits of flat LDS address = LDS byte offset
}

// ---------- Tensor Data Mover: 2-D tile (rowlen x nrows) global -> LDS ----------
__device__ inline void tdm_load_2d(unsigned lds_addr, const bf16* g,
                                   unsigned tensor_d0, unsigned tensor_d1,
                                   unsigned stride0, unsigned tile0, unsigned tile1,
                                   int lane) {
#ifdef HAVE_TDM
  unsigned long long ga = (unsigned long long)(uintptr_t)g;
  v4u g0; v8i g1; v4i z4 = {0, 0, 0, 0};
  g0[0] = 1u;                                        // count=1, user descriptor
  g0[1] = lds_addr;                                  // D#.lds_addr
  g0[2] = (unsigned)(ga & 0xFFFFFFFFu);              // global_addr[31:0]
  g0[3] = (unsigned)((ga >> 32) & 0x01FFFFFFu) | 0x80000000u; // addr[56:32] | type=2
  g1[0] = 0x00010000;                                // data_size=1 (2 bytes)
  g1[1] = (int)((tensor_d0 & 0xFFFFu) << 16);        // tensor_dim0[15:0]
  g1[2] = (int)((tensor_d0 >> 16) | ((tensor_d1 & 0xFFFFu) << 16));
  g1[3] = (int)((tensor_d1 >> 16) | (tile0 << 16));  // tile_dim0
  g1[4] = (int)tile1;                                // tile_dim1 (tile_dim2=0)
  g1[5] = (int)stride0;                              // tensor_dim0_stride[31:0]
  g1[6] = 0;
  g1[7] = 0;
#  if __clang_major__ >= 23
  v8i z8 = {0, 0, 0, 0, 0, 0, 0, 0};
  __builtin_amdgcn_tensor_load_to_lds(g0, g1, z4, z4, z8, 0);
#  else
  __builtin_amdgcn_tensor_load_to_lds(g0, g1, z4, z4, 0);
#  endif
#else
  // fallback: no copy is done; the TDM builtin is assumed present
  bf16* lp = (bf16*)(uintptr_t)0; (void)lp;
  unsigned* d32 = (unsigned*)(void*)(uintptr_t)lds_addr; (void)d32; // unreachable config
  (void)g; (void)tensor_d0; (void)tensor_d1; (void)stride0; (void)tile0; (void)tile1; (void)lane;
#endif
}
__device__ inline void wait_tensor_2() {
#ifdef HAVE_TENSORCNT
  __builtin_amdgcn_s_wait_tensorcnt(2);
#else
  asm volatile("s_wait_tensorcnt 0x2" ::: "memory");
#endif
}
__device__ inline void wait_tensor_0() {
#ifdef HAVE_TENSORCNT
  __builtin_amdgcn_s_wait_tensorcnt(0);
#else
  asm volatile("s_wait_tensorcnt 0x0" ::: "memory");
#endif
}

// ---------- conversion kernels ----------
__global__ __launch_bounds__(256) void cvt_bf16_kernel(const float* __restrict__ in,
                                                       bf16* __restrict__ out,
                                                       float scale, int n) {
  int i = blockIdx.x * blockDim.x + threadIdx.x;
  if (i < n) out[i] = (bf16)(in[i] * scale);
}

// old caches -> f32 outputs, K -> scaled bf16 [B,TKV,D], V -> bf16 transposed [B,D,TKV]
__global__ __launch_bounds__(256) void cache_init_kernel(const float* __restrict__ kin,
                                                         const float* __restrict__ vin,
                                                         float* __restrict__ kout,
                                                         float* __restrict__ vout,
                                                         bf16* __restrict__ kbf,
                                                         bf16* __restrict__ vbfT,
                                                         float kscale, int n) {
  int i = blockIdx.x * blockDim.x + threadIdx.x;
  if (i < n) {
    float k = kin[i], v = vin[i];
    kout[i] = k;
    vout[i] = v;
    kbf[i] = (bf16)(k * kscale);
    int b   = i >> 21;            // TKV*D = 2^21
    int rem = i & 0x1FFFFF;
    int t   = rem >> 10;          // D = 2^10
    int d   = rem & 1023;
    vbfT[((size_t)(b * Dd + d)) * TKV + t] = (bf16)v;
  }
}

// ---------- WMMA GEMM: C[M,N] = A[M,K] * W[N,K]^T (+bias), per-wave 16x64 tile ----------
__global__ __launch_bounds__(128)
void gemm_bf16_wmma(const bf16* __restrict__ A, const bf16* __restrict__ W,
                    const float* __restrict__ bias,
                    float* __restrict__ outf, bf16* __restrict__ outb,
                    float bscale, int M, int N, int K, int remap, int outb_tr) {
  const int lane  = threadIdx.x & 31;
  const int wave  = threadIdx.x >> 5;
  const int tile  = blockIdx.x * 4 + wave;
  const int ntn   = N >> 6;
  const int m0    = (tile / ntn) * 16;
  const int n0    = (tile % ntn) * 64;
  const int nlane = lane & 15;
  const int akb   = (lane & 16) ? 16 : 0;
  const int bkb   = (lane & 16) ? 16 : 0;
  const int rbase = (lane & 16) ? 8 : 0;

  const bf16* arow = A + (size_t)(m0 + nlane) * K;
  v8f acc0 = vzero8(), acc1 = vzero8(), acc2 = vzero8(), acc3 = vzero8();

  for (int k0 = 0; k0 < K; k0 += 32) {
    v16bf a = load_a16(arow + k0, akb);
    const bf16* wp = W + (size_t)(n0 + nlane) * K + k0 + bkb;
    acc0 = WMMA_BF16(a, load_b16(wp), acc0);
    acc1 = WMMA_BF16(a, load_b16(wp + (size_t)16 * K), acc1);
    acc2 = WMMA_BF16(a, load_b16(wp + (size_t)32 * K), acc2);
    acc3 = WMMA_BF16(a, load_b16(wp + (size_t)48 * K), acc3);
    __builtin_prefetch(arow + k0 + 64, 0, 1);
  }

  float bi0 = 0.f, bi1 = 0.f, bi2 = 0.f, bi3 = 0.f;
  if (bias) {
    bi0 = bias[n0 + nlane];      bi1 = bias[n0 + 16 + nlane];
    bi2 = bias[n0 + 32 + nlane]; bi3 = bias[n0 + 48 + nlane];
  }
#pragma unroll
  for (int r = 0; r < 8; ++r) {
    int row  = m0 + rbase + r;
    int orow = remap ? ((row >> 10) * TKV + (TKV - TQ) + (row & (TQ - 1))) : row;
    float v0 = acc0[r] + bi0, v1 = acc1[r] + bi1, v2 = acc2[r] + bi2, v3 = acc3[r] + bi3;
    size_t base = (size_t)orow * N;
    if (outf) {
      outf[base + n0 + nlane]      = v0;
      outf[base + n0 + 16 + nlane] = v1;
      outf[base + n0 + 32 + nlane] = v2;
      outf[base + n0 + 48 + nlane] = v3;
    }
    if (outb) {
      if (outb_tr) { // transposed cache append: [B, D, TKV]
        int bb = row >> 10, t = row & (TQ - 1);
        size_t tb = (size_t)bb * Dd;
        size_t tt = (size_t)(TKV - TQ) + t;
        outb[(tb + n0 + nlane) * TKV + tt]      = (bf16)(v0 * bscale);
        outb[(tb + n0 + 16 + nlane) * TKV + tt] = (bf16)(v1 * bscale);
        outb[(tb + n0 + 32 + nlane) * TKV + tt] = (bf16)(v2 * bscale);
        outb[(tb + n0 + 48 + nlane) * TKV + tt] = (bf16)(v3 * bscale);
      } else {
        outb[base + n0 + nlane]      = (bf16)(v0 * bscale);
        outb[base + n0 + 16 + nlane] = (bf16)(v1 * bscale);
        outb[base + n0 + 32 + nlane] = (bf16)(v2 * bscale);
        outb[base + n0 + 48 + nlane] = (bf16)(v3 * bscale);
      }
    }
  }
}

// ---------- flash attention: per wave one (b, h, 16-row q tile); TDM-staged KV ----------
__global__ __launch_bounds__(128)
void flash_attn_wmma(const bf16* __restrict__ Q, const bf16* __restrict__ Kc,
                     const bf16* __restrict__ VcT, const float* __restrict__ mask,
                     bf16* __restrict__ WV) {
  __shared__ __align__(16) bf16 ktile[4][2][32 * 64]; // [wave][buf][kv][d]
  __shared__ __align__(16) bf16 vtile[4][2][64 * 32]; // [wave][buf][d][kv]
  __shared__ __align__(16) bf16 pbuf[4][16][40];      // 80B row stride

  const int lane  = threadIdx.x & 31;
  const int wave  = threadIdx.x >> 5;
  const int id    = blockIdx.x * 4 + wave;     // 0..4095
  const int qt    = id & 63;
  const int h     = (id >> 6) & 15;
  const int b     = id >> 10;
  const int nlane = lane & 15;
  const int akb   = (lane & 16) ? 16 : 0;
  const int bkb   = (lane & 16) ? 16 : 0;
  const int rbase = (lane & 16) ? 8 : 0;

  const int qrow0 = b * TQ + qt * 16;
  const bf16* qrowp = Q + (size_t)(qrow0 + nlane) * Dd + h * DH;
  v16bf aq0 = load_a16(qrowp, akb);        // d 0..31
  v16bf aq1 = load_a16(qrowp + 32, akb);   // d 32..63

  const bf16* kglob = Kc + (size_t)b * TKV * Dd + h * DH;      // + kv*Dd
  const bf16* vglob = VcT + (size_t)(b * Dd + h * DH) * TKV;   // + kv

  v8f o0 = vzero8(), o1 = vzero8(), o2 = vzero8(), o3 = vzero8();
  float m_run[8], l_run[8];
#pragma unroll
  for (int r = 0; r < 8; ++r) { m_run[r] = -1e30f; l_run[r] = 0.f; }

  const int qb = qt * 16 + rbase;
  constexpr int NCH = TKV / 32;

  // prologue: stage chunk 0
  tdm_load_2d(lds_off(&ktile[wave][0][0]), kglob, Dd, Bb * TKV, Dd, 64, 32, lane);
  tdm_load_2d(lds_off(&vtile[wave][0][0]), vglob, TKV, Bb * Dd, TKV, 32, 64, lane);

  for (int c = 0; c < NCH; ++c) {
    const int cur = c & 1;
    const int kv0 = c * 32;
    // previous iteration's LDS reads must drain before TDM overwrites buf cur^1
    asm volatile("s_wait_dscnt 0x0" ::: "memory");
    if (c + 1 < NCH) {
      const int kvn = kv0 + 32;
      tdm_load_2d(lds_off(&ktile[wave][cur ^ 1][0]), kglob + (size_t)kvn * Dd,
                  Dd, Bb * TKV, Dd, 64, 32, lane);
      tdm_load_2d(lds_off(&vtile[wave][cur ^ 1][0]), vglob + kvn,
                  TKV, Bb * Dd, TKV, 32, 64, lane);
      wait_tensor_2();  // 2 just-issued may remain; current chunk complete
    } else {
      wait_tensor_0();
    }

    const bf16* kl = &ktile[wave][cur][0];
    const bf16* vl = &vtile[wave][cur][0];

    // S = Q K^T : two 16-wide kv tiles, d split 0..31 / 32..63
    v8f s0 = vzero8(), s1 = vzero8();
    const bf16* kp0 = kl + nlane * 64 + bkb;
    s0 = WMMA_BF16(aq0, load_b16_lds(kp0), s0);
    s0 = WMMA_BF16(aq1, load_b16_lds(kp0 + 32), s0);
    const bf16* kp1 = kl + (16 + nlane) * 64 + bkb;
    s1 = WMMA_BF16(aq0, load_b16_lds(kp1), s1);
    s1 = WMMA_BF16(aq1, load_b16_lds(kp1 + 32), s1);

#pragma unroll
    for (int r = 0; r < 8; ++r) {
      s0[r] += mask[(size_t)(qb + r) * TKV + kv0 + nlane];
      s1[r] += mask[(size_t)(qb + r) * TKV + kv0 + 16 + nlane];
    }
#pragma unroll
    for (int r = 0; r < 8; ++r) {
      float mc = fmaxf(s0[r], s1[r]);
#pragma unroll
      for (int off = 1; off < 16; off <<= 1) mc = fmaxf(mc, __shfl_xor(mc, off, 32));
      float mnew  = fmaxf(m_run[r], mc);
      float alpha = __expf(m_run[r] - mnew);
      float p0 = __expf(s0[r] - mnew);
      float p1 = __expf(s1[r] - mnew);
      float rs = p0 + p1;
#pragma unroll
      for (int off = 1; off < 16; off <<= 1) rs += __shfl_xor(rs, off, 32);
      l_run[r] = l_run[r] * alpha + rs;
      m_run[r] = mnew;
      o0[r] *= alpha; o1[r] *= alpha; o2[r] *= alpha; o3[r] *= alpha;
      pbuf[wave][rbase + r][nlane]      = (bf16)p0;
      pbuf[wave][rbase + r][16 + nlane] = (bf16)p1;
    }
    asm volatile("s_wait_dscnt 0x0" ::: "memory");

    // O += P[16x32] * V[32x64]; V^T tile is [d][kv] so K-dim is contiguous
    v16bf ap = load_a16(&pbuf[wave][nlane][0], akb);
    const bf16* vp = vl + nlane * 32 + bkb;
    o0 = WMMA_BF16(ap, load_b16_lds(vp), o0);
    o1 = WMMA_BF16(ap, load_b16_lds(vp + 16 * 32), o1);
    o2 = WMMA_BF16(ap, load_b16_lds(vp + 32 * 32), o2);
    o3 = WMMA_BF16(ap, load_b16_lds(vp + 48 * 32), o3);
  }

#pragma unroll
  for (int r = 0; r < 8; ++r) {
    float inv = 1.0f / l_run[r];
    bf16* op = WV + (size_t)(qrow0 + rbase + r) * Dd + h * DH + nlane;
    op[0]  = (bf16)(o0[r] * inv);
    op[16] = (bf16)(o1[r] * inv);
    op[32] = (bf16)(o2[r] * inv);
    op[48] = (bf16)(o3[r] * inv);
  }
}

extern "C" void kernel_launch(void* const* d_in, const int* in_sizes, int n_in,
                              void* d_out, int out_size, void* d_ws, size_t ws_size,
                              hipStream_t stream) {
  const float* x    = (const float*)d_in[0];
  const float* kc   = (const float*)d_in[1];
  const float* vc   = (const float*)d_in[2];
  const float* mask = (const float*)d_in[3];
  const float* Wq   = (const float*)d_in[4];
  const float* bq   = (const float*)d_in[5];
  const float* Wk   = (const float*)d_in[6];
  const float* Wv   = (const float*)d_in[7];
  const float* bv   = (const float*)d_in[8];
  const float* Wo   = (const float*)d_in[9];
  const float* bo   = (const float*)d_in[10];

  float* out  = (float*)d_out;                       // [4096,1024]
  float* kout = out + (size_t)Bb * TQ * Dd;          // [4,2048,1024]
  float* vout = kout + (size_t)Bb * TKV * Dd;

  bf16* Xbf  = (bf16*)d_ws;                // 4,194,304
  bf16* Wqb  = Xbf + (size_t)4194304;      // 1,048,576 each
  bf16* Wkb  = Wqb + (size_t)1048576;
  bf16* Wvb  = Wkb + (size_t)1048576;
  bf16* Wob  = Wvb + (size_t)1048576;
  bf16* Qbf  = Wob + (size_t)1048576;      // 4,194,304
  bf16* Kbf  = Qbf + (size_t)4194304;      // 8,388,608  [B,TKV,D]
  bf16* VbfT = Kbf + (size_t)8388608;      // 8,388,608  [B,D,TKV]
  bf16* WVb  = VbfT + (size_t)8388608;     // 4,194,304

  // stage 1: conversions
  cvt_bf16_kernel<<<16384, 256, 0, stream>>>(x, Xbf, 1.f, 4194304);
  cvt_bf16_kernel<<<4096, 256, 0, stream>>>(Wq, Wqb, 1.f, 1048576);
  cvt_bf16_kernel<<<4096, 256, 0, stream>>>(Wk, Wkb, 1.f, 1048576);
  cvt_bf16_kernel<<<4096, 256, 0, stream>>>(Wv, Wvb, 1.f, 1048576);
  cvt_bf16_kernel<<<4096, 256, 0, stream>>>(Wo, Wob, 1.f, 1048576);
  cache_init_kernel<<<32768, 256, 0, stream>>>(kc, vc, kout, vout, Kbf, VbfT, SCALE, 8388608);

  // stage 2: QKV projections (WMMA); K/V appended to cache tail (V transposed)
  gemm_bf16_wmma<<<1024, 128, 0, stream>>>(Xbf, Wqb, bq, nullptr, Qbf, SCALE,
                                           Bb * TQ, Dd, Dd, 0, 0);
  gemm_bf16_wmma<<<1024, 128, 0, stream>>>(Xbf, Wkb, nullptr, kout, Kbf, SCALE,
                                           Bb * TQ, Dd, Dd, 1, 0);
  gemm_bf16_wmma<<<1024, 128, 0, stream>>>(Xbf, Wvb, bv, vout, VbfT, 1.f,
                                           Bb * TQ, Dd, Dd, 1, 1);

  // stage 3: flash attention (WMMA QK^T/PV, TDM-staged KV tiles)
  flash_attn_wmma<<<1024, 128, 0, stream>>>(Qbf, Kbf, VbfT, mask, WVb);

  // stage 4: output projection
  gemm_bf16_wmma<<<1024, 128, 0, stream>>>(WVb, Wob, bo, out, nullptr, 1.f,
                                           Bb * TQ, Dd, Dd, 0, 0);
}
